// SelfAttention_43757126812339
// MI455X (gfx1250) — compile-verified, matrix operand order checked by reading, not run on hardware
//
#include <hip/hip_runtime.h>
#include <stdint.h>

// ---------------------------------------------------------------------------
// Self-attention with summed output, MI455X (gfx1250, wave32, WMMA bf16).
//   attention = feats @ W1 + b1
//   scores    = attention @ feats^T   (per batch)
//   P         = softmax(scores, axis=-1)
//   out[b,h]  = sum_n (P @ feats)[b,n,h] = sum_m colsum[b,m] * feats[b,m,h]
// where colsum[b,m] = sum_n P[b,n,m].  This removes the second big einsum.
// Register-blocked WMMA: 8 WMMAs / 6 frag loads (fc1), 4 / 5 (scores).
// ---------------------------------------------------------------------------

typedef __attribute__((ext_vector_type(16))) __bf16 v16bf;
typedef __attribute__((ext_vector_type(8)))  float  v8f;

#define HDIM 1024
#define NSEQ 2048
#define BATCH 16

__device__ __forceinline__ __bf16 f2bf(float x) {
  union { float f; uint32_t u; } a; a.f = x;
  uint32_t r = a.u + 0x7FFFu + ((a.u >> 16) & 1u);   // round-to-nearest-even
  union { uint16_t s; __bf16 b; } c; c.s = (uint16_t)(r >> 16);
  return c.b;
}

union FragU { v16bf v; uint4 q[2]; };

// Load one 16x32 bf16 A/B fragment (wave32 layout per CDNA5 ISA 7.12.2):
// lane<16 : row=lane,    K = {k0+0..7} in v0..3, {k0+16..23} in v4..7
// lane>=16: row=lane-16, K = {k0+8..15},          {k0+24..31}
// `base` is row-major [16 x ld] bf16. For B of A*B^T, store B as [col][K] and
// load identically.
__device__ __forceinline__ v16bf load_frag16(const __bf16* __restrict__ base,
                                             int ld, int k0) {
  int lane = threadIdx.x & 31;
  int row  = lane & 15;
  int koff = (lane & 16) ? 8 : 0;
  const __bf16* p = base + (size_t)row * ld + k0 + koff;
  FragU f;
  f.q[0] = *(const uint4*)(p);        // 8 bf16
  f.q[1] = *(const uint4*)(p + 16);   // 8 bf16
  return f.v;
}

#define WMMA_BF16(A, B, C) \
  __builtin_amdgcn_wmma_f32_16x16x32_bf16(false, (A), false, (B), (short)0, (C), false, false)

// ---------------- kernel 1a: feats f32 -> bf16 (vectorized) ----------------
__global__ void k_convert(const float* __restrict__ in,
                          __bf16* __restrict__ out, int n8) {
  int i = blockIdx.x * blockDim.x + threadIdx.x;
  if (i >= n8) return;
  const float4* p = (const float4*)in + (size_t)i * 2;
  float4 x0 = p[0], x1 = p[1];
  union { __bf16 b[8]; uint4 q; } o;
  o.b[0]=f2bf(x0.x); o.b[1]=f2bf(x0.y); o.b[2]=f2bf(x0.z); o.b[3]=f2bf(x0.w);
  o.b[4]=f2bf(x1.x); o.b[5]=f2bf(x1.y); o.b[6]=f2bf(x1.z); o.b[7]=f2bf(x1.w);
  ((uint4*)out)[i] = o.q;
}

// ------------- kernel 1b: W1T[h][k] = bf16(W1[k][h]) (LDS tiled) -----------
__global__ void k_transpose_w1(const float* __restrict__ W1,
                               __bf16* __restrict__ W1T) {
  __shared__ float tile[32][33];
  int bx = blockIdx.x * 32, by = blockIdx.y * 32;
  int tx = threadIdx.x & 31, ty = threadIdx.x >> 5;   // 32 x 8
  #pragma unroll
  for (int i = 0; i < 32; i += 8)
    tile[ty + i][tx] = W1[(size_t)(by + ty + i) * HDIM + bx + tx];
  __syncthreads();
  #pragma unroll
  for (int i = 0; i < 32; i += 8)
    W1T[(size_t)(bx + ty + i) * HDIM + by + tx] = f2bf(tile[tx][ty + i]);
}

// ---------------- kernel 1c: zero colsum accumulator -----------------------
__global__ void k_zero(float* __restrict__ p, int n) {
  int i = blockIdx.x * blockDim.x + threadIdx.x;
  if (i < n) p[i] = 0.f;
}

// ---------------- kernel 2: attention = feats @ W1 + b1 (bf16 WMMA) --------
// Each wave computes a 32x64 output tile: 2 A-frags x 4 B-frags = 8 WMMAs
// per K-step from 6 fragment loads. Waves in a block share the A strip.
// grid 2048 blocks x 256 threads (8 waves); wave tiles: 1024 (M) x 16 (N).
__global__ void k_fc1(const __bf16* __restrict__ feats_bf,
                      const __bf16* __restrict__ W1T,
                      const float* __restrict__ b1,
                      __bf16* __restrict__ attn_bf) {
  int wid  = blockIdx.x * 8 + (threadIdx.x >> 5);
  int mi   = wid >> 4;                 // 0..1023, 32-row tiles
  int ni   = wid & 15;                 // 0..15,   64-col tiles
  int row0 = mi * 32;
  int h0   = ni * 64;
  const __bf16* A0 = feats_bf + (size_t)row0 * HDIM;
  const __bf16* B0 = W1T      + (size_t)h0   * HDIM;

  v8f acc[2][4] = {};
  #pragma unroll 2
  for (int kc = 0; kc < HDIM; kc += 32) {
    v16bf a0 = load_frag16(A0,             HDIM, kc);
    v16bf a1 = load_frag16(A0 + 16 * HDIM, HDIM, kc);
    v16bf bb0 = load_frag16(B0,             HDIM, kc);
    v16bf bb1 = load_frag16(B0 + 16 * HDIM, HDIM, kc);
    v16bf bb2 = load_frag16(B0 + 32 * HDIM, HDIM, kc);
    v16bf bb3 = load_frag16(B0 + 48 * HDIM, HDIM, kc);
    acc[0][0] = WMMA_BF16(a0, bb0, acc[0][0]);
    acc[0][1] = WMMA_BF16(a0, bb1, acc[0][1]);
    acc[0][2] = WMMA_BF16(a0, bb2, acc[0][2]);
    acc[0][3] = WMMA_BF16(a0, bb3, acc[0][3]);
    acc[1][0] = WMMA_BF16(a1, bb0, acc[1][0]);
    acc[1][1] = WMMA_BF16(a1, bb1, acc[1][1]);
    acc[1][2] = WMMA_BF16(a1, bb2, acc[1][2]);
    acc[1][3] = WMMA_BF16(a1, bb3, acc[1][3]);
  }

  // C layout: vgpr r -> M = r + (lane>=16 ? 8 : 0); N = lane&15
  int lane = threadIdx.x & 31;
  #pragma unroll
  for (int bi = 0; bi < 4; ++bi) {
    int col = h0 + bi * 16 + (lane & 15);
    float bias = b1[col];
    #pragma unroll
    for (int ai = 0; ai < 2; ++ai) {
      int rbase = row0 + ai * 16 + ((lane & 16) ? 8 : 0);
      #pragma unroll
      for (int r = 0; r < 8; ++r)
        attn_bf[(size_t)(rbase + r) * HDIM + col] = f2bf(acc[ai][bi][r] + bias);
    }
  }
}

// --------- kernel 3: scores row-tile + softmax + column-sum ---------------
// grid (16*128), block 256, dyn LDS = (16*2048 + 256 + 32) floats.
__global__ void k_attn(const __bf16* __restrict__ attn_bf,
                       const __bf16* __restrict__ feats_bf,
                       float* __restrict__ colsum) {
  extern __shared__ float smem[];
  float* S     = smem;                  // [16][2048] raw scores
  float* red   = smem + 16 * NSEQ;      // [256] reduction scratch
  float* stats = red + 256;             // [0..15]=rowmax, [16..31]=1/rowsum

  int b    = blockIdx.x >> 7;
  int n0   = (blockIdx.x & 127) * 16;
  int wave = threadIdx.x >> 5;
  int lane = threadIdx.x & 31;
  const __bf16* Aq = attn_bf  + ((size_t)b * NSEQ + n0) * HDIM;
  const __bf16* Kf = feats_bf + (size_t)b * NSEQ * HDIM;

  // ---- scores GEMM: each wave handles 4 groups of 4 key tiles ----
  // (one A-frag load feeds four WMMAs: 4 WMMAs / 5 loads)
  for (int mq = wave; mq < NSEQ / 64; mq += 8) {
    int m0 = mq * 64;
    const __bf16* K0 = Kf + (size_t)m0 * HDIM;
    v8f acc[4] = {};
    #pragma unroll 2
    for (int kc = 0; kc < HDIM; kc += 32) {
      v16bf a  = load_frag16(Aq, HDIM, kc);
      v16bf k0 = load_frag16(K0,             HDIM, kc);
      v16bf k1 = load_frag16(K0 + 16 * HDIM, HDIM, kc);
      v16bf k2 = load_frag16(K0 + 32 * HDIM, HDIM, kc);
      v16bf k3 = load_frag16(K0 + 48 * HDIM, HDIM, kc);
      acc[0] = WMMA_BF16(a, k0, acc[0]);
      acc[1] = WMMA_BF16(a, k1, acc[1]);
      acc[2] = WMMA_BF16(a, k2, acc[2]);
      acc[3] = WMMA_BF16(a, k3, acc[3]);
    }
    int col = m0 + (lane & 15);
    int rb  = (lane & 16) ? 8 : 0;
    #pragma unroll
    for (int t = 0; t < 4; ++t)
      #pragma unroll
      for (int r = 0; r < 8; ++r)
        S[(rb + r) * NSEQ + col + t * 16] = acc[t][r];
  }
  __syncthreads();

  // ---- softmax stats: 16 threads per row ----
  int row = threadIdx.x >> 4;           // 0..15
  int j   = threadIdx.x & 15;
  float mloc = -3.4e38f;
  for (int c = j; c < NSEQ; c += 16) mloc = fmaxf(mloc, S[row * NSEQ + c]);
  red[threadIdx.x] = mloc; __syncthreads();
  for (int s = 8; s > 0; s >>= 1) {
    if (j < s) red[threadIdx.x] = fmaxf(red[threadIdx.x], red[threadIdx.x + s]);
    __syncthreads();
  }
  float rmax = red[row << 4];
  __syncthreads();
  float sloc = 0.f;
  for (int c = j; c < NSEQ; c += 16) sloc += __expf(S[row * NSEQ + c] - rmax);
  red[threadIdx.x] = sloc; __syncthreads();
  for (int s = 8; s > 0; s >>= 1) {
    if (j < s) red[threadIdx.x] += red[threadIdx.x + s];
    __syncthreads();
  }
  if (j == 0) { stats[row] = rmax; stats[16 + row] = 1.f / red[row << 4]; }
  __syncthreads();

  // ---- column partial sums over this 16-row tile -> global colsum ----
  for (int c = threadIdx.x; c < NSEQ; c += 256) {
    float acc = 0.f;
    #pragma unroll
    for (int r = 0; r < 16; ++r)
      acc += __expf(S[r * NSEQ + c] - stats[r]) * stats[16 + r];
    atomicAdd(&colsum[(size_t)b * NSEQ + c], acc);
  }
}

// ------- kernel 4: out[b,h] = sum_m colsum[b,m] * feats[b,m,h] (fp32) ------
__global__ void k_final(const float* __restrict__ feats,
                        const float* __restrict__ colsum,
                        float* __restrict__ out) {
  int b = blockIdx.x;
  int h = blockIdx.y * 256 + threadIdx.x;
  const float* F  = feats  + (size_t)b * NSEQ * HDIM + h;
  const float* cs = colsum + (size_t)b * NSEQ;
  float acc = 0.f;
  #pragma unroll 4
  for (int m = 0; m < NSEQ; ++m) acc += cs[m] * F[(size_t)m * HDIM];
  out[b * HDIM + h] = acc;
}

// ---------------------------------------------------------------------------
extern "C" void kernel_launch(void* const* d_in, const int* in_sizes, int n_in,
                              void* d_out, int out_size, void* d_ws, size_t ws_size,
                              hipStream_t stream) {
  const float* feats = (const float*)d_in[0];   // [16,2048,1024]
  const float* W1    = (const float*)d_in[1];   // [1024,1024]
  const float* b1    = (const float*)d_in[2];   // [1024]
  float* out = (float*)d_out;                   // [16,1024]

  // workspace layout
  char* ws = (char*)d_ws;
  const size_t n_feats = (size_t)BATCH * NSEQ * HDIM;          // 33.5M elems
  __bf16* feats_bf = (__bf16*)ws;                              // 64 MB
  __bf16* attn_bf  = (__bf16*)(ws + n_feats * 2);              // 64 MB
  __bf16* W1T      = (__bf16*)(ws + n_feats * 4);              // 2 MB
  float*  colsum   = (float*) (ws + n_feats * 4 + (size_t)HDIM * HDIM * 2);

  // 1) precision conversion / transpose / zero accumulators
  {
    int n8 = (int)(n_feats / 8);
    k_convert<<<dim3((n8 + 255) / 256), dim3(256), 0, stream>>>(feats, feats_bf, n8);
    k_transpose_w1<<<dim3(HDIM / 32, HDIM / 32), dim3(256), 0, stream>>>(W1, W1T);
    int nc = BATCH * NSEQ;
    k_zero<<<dim3((nc + 255) / 256), dim3(256), 0, stream>>>(colsum, nc);
  }

  // 2) fc1 GEMM: [32768 x 1024] x [1024 x 1024] in bf16 WMMA,
  //    32x64 tile per wave -> 16384 wave tiles / 8 waves per block.
  k_fc1<<<dim3(2048), dim3(256), 0, stream>>>(feats_bf, W1T, b1, attn_bf);

  // 3) fused scores + softmax + column-sum
  size_t smem = (size_t)(16 * NSEQ + 256 + 32) * sizeof(float);   // ~129 KB LDS
  k_attn<<<dim3(BATCH * (NSEQ / 16)), dim3(256), smem, stream>>>(
      attn_bf, feats_bf, colsum);

  // 4) out[b,h] = colsum . feats  (fp32, bandwidth-trivial)
  k_final<<<dim3(BATCH, HDIM / 256), dim3(256), 0, stream>>>(feats, colsum, out);
}
